// SimpleRNN_50087908606436
// MI455X (gfx1250) — matrix operand order repaired and checked by reading, not compiled, not run on hardware
//
#include <hip/hip_runtime.h>
#include <math.h>
#include <stdint.h>

#define T_STEPS 512
#define BATCH   2048
#define IN_DIM  28
#define OUT_DIM 10
#define HID     198

#define KCH   7            // 7 K-chunks of 32 -> K padded to 224
#define NPAD  208          // 13 N-tiles of 16 (one per wave)
#define HSTR  232          // bf16 LDS row stride (464B: 16B-aligned, conflict-free)
#define FSTR  208          // fp32 final-h row stride
#define XSTR  36           // x LDS row stride in floats (144B: conflict-free, room for tail)

typedef __attribute__((ext_vector_type(16))) __bf16 v16bf;
typedef __attribute__((ext_vector_type(8)))  __bf16 v8bf;
typedef __attribute__((ext_vector_type(8)))  float  v8f;
typedef __attribute__((ext_vector_type(4)))  float  f4;

union V16 { v16bf v; v8bf h[2]; };

#if __has_builtin(__builtin_amdgcn_tanhf)
#define TANHF(x) __builtin_amdgcn_tanhf(x)
#elif __has_builtin(__builtin_amdgcn_tanh_f32)
#define TANHF(x) __builtin_amdgcn_tanh_f32(x)
#else
#define TANHF(x) tanhf(x)
#endif

#define WMMA(a, b, c) \
  __builtin_amdgcn_wmma_f32_16x16x32_bf16(false, (a), false, (b), (short)0, (c), false, false)

// waves 0..3: async-copy the 16x28-float x tile for step tn into an LDS buffer.
// 112 lanes x 16B = 1792B; per-lane LDS address carries the row padding.
__device__ __forceinline__ void prefetch_x(const float* __restrict__ x, float* dst,
                                           long tn, int b0, int w, int lane) {
  if (w < 4) {
    int cid = w * 32 + lane;
    if (cid < 112) {
      int m = cid / 7, q = cid - m * 7;
      const float* gp = x + (tn * BATCH + b0 + m) * IN_DIM + q * 4;
      unsigned int lp = (unsigned int)(uintptr_t)(dst + m * XSTR + q * 4);
      asm volatile("global_load_async_to_lds_b128 %0, %1, off"
                   :: "v"(lp), "v"((unsigned long long)(uintptr_t)gp)
                   : "memory");
    }
  }
}

__global__ __launch_bounds__(416) void rnn_persist(
    const float* __restrict__ x,  const float* __restrict__ H,
    const float* __restrict__ U,  const float* __restrict__ A,
    const float* __restrict__ bv, const float* __restrict__ cv,
    float* __restrict__ out)
{
  // h kept as split bf16 (hi + lo); H kept as hi (VGPRs) + lo (LDS).
  __shared__ __attribute__((aligned(16))) __bf16 hhi[16 * HSTR];    //  7.4 KB
  __shared__ __attribute__((aligned(16))) __bf16 hlo[16 * HSTR];    //  7.4 KB
  __shared__ __attribute__((aligned(16))) float  hf [16 * FSTR];    // 13.3 KB
  __shared__ __attribute__((aligned(16))) __bf16 Hlo[NPAD * HSTR];  // 96.5 KB
  __shared__ __attribute__((aligned(16))) float  xbuf[2][16 * XSTR]; // 4.6 KB

  const int tid  = threadIdx.x;
  const int w    = tid >> 5;          // wave id == N-tile id (0..12)
  const int lane = tid & 31;
  const int lo16 = lane & 15;
  const int hi   = lane >> 4;         // half-wave select
  const int b0   = blockIdx.x * 16;   // batch tile base

  const int j  = w * 16 + lo16;       // N index: H row / U column / bias index
  const int kb = hi * 16;             // unused; B-frag uses the A-frag K order (ka)
  const int ka = hi * 8;              // A-frag: elems0..7 -> base+ka ; 8..15 -> base+16+ka

  // ---- persistent B hi-operands in VGPRs: H^T tile + U^T tile (56+16 VGPRs)
  v16bf Hbh[KCH];
#pragma unroll
  for (int c = 0; c < KCH; ++c) {
    V16 vh;
#pragma unroll
    for (int e = 0; e < 16; ++e) {
      int k  = c * 32 + (e < 8 ? ka + e : 16 + ka + (e - 8));
      float f = (j < HID && k < HID) ? H[j * HID + k] : 0.0f;
      vh.v[e] = (__bf16)f;
    }
    Hbh[c] = vh.v;
  }
  v16bf Ubh, Ubl;
  {
    V16 vh, vl;
#pragma unroll
    for (int e = 0; e < 16; ++e) {
      int k  = (e < 8 ? ka + e : 16 + ka + (e - 8));
      float f = (k < IN_DIM && j < HID) ? U[(long)k * HID + j] : 0.0f;
      __bf16 fh = (__bf16)f;
      vh.v[e] = fh;
      vl.v[e] = (__bf16)(f - (float)fh);
    }
    Ubh = vh.v; Ubl = vl.v;
  }
  const float bj = (j < HID) ? bv[j] : 0.0f;

  // ---- fill LDS: H_lo residual, h0 = 0; async-prefetch x tile for t=0
  for (int idx = tid; idx < NPAD * 224; idx += 416) {
    int r = idx / 224, cc = idx % 224;
    float f = (r < HID && cc < HID) ? H[r * HID + cc] : 0.0f;
    __bf16 fh = (__bf16)f;
    Hlo[r * HSTR + cc] = (__bf16)(f - (float)fh);
  }
  for (int i = tid; i < 16 * HSTR; i += 416) {
    hhi[i] = (__bf16)0.0f; hlo[i] = (__bf16)0.0f;
  }
  prefetch_x(x, xbuf[0], 0, b0, w, lane);
  asm volatile("s_wait_asynccnt 0x0" ::: "memory");
  __syncthreads();

  const int jrow = j * HSTR;          // base of this lane's H_lo row
  float nh[8];

  // ---- recurrent scan (unrolled ONCE: 13 waves -> <=256 VGPRs/wave, no spills)
#pragma unroll 1
  for (int t = 0; t < T_STEPS; ++t) {
    // kick async copy of next x tile into the other buffer (clamped: branch-free)
    long tn = (t + 1 < T_STEPS) ? (long)(t + 1) : (long)(T_STEPS - 1);
    prefetch_x(x, xbuf[(t + 1) & 1], tn, b0, w, lane);

    v8f acc = {};

    // fused xu_t tile from LDS (prefetched last step): K padded 28->32
    {
      const float* xb = &xbuf[t & 1][lo16 * XSTR];
      f4 z  = {};
      f4 f0 = *(const f4*)(xb + ka);
      f4 f1 = *(const f4*)(xb + ka + 4);
      f4 f2 = *(const f4*)(xb + 16 + ka);   // hi=0: 16..19, hi=1: 24..27
      f4 f3 = *(const f4*)(xb + 20);        // always in-bounds (padded row)
      if (hi) f3 = z;                       // K 28..31 pad -> cndmask, no branch
      float fe[16] = { f0[0],f0[1],f0[2],f0[3], f1[0],f1[1],f1[2],f1[3],
                       f2[0],f2[1],f2[2],f2[3], f3[0],f3[1],f3[2],f3[3] };
      V16 xh, xl;
#pragma unroll
      for (int e = 0; e < 16; ++e) {
        __bf16 fh = (__bf16)fe[e];
        xh.v[e] = fh;
        xl.v[e] = (__bf16)(fe[e] - (float)fh);
      }
      acc = WMMA(xh.v, Ubh, acc);
      acc = WMMA(xh.v, Ubl, acc);
      acc = WMMA(xl.v, Ubh, acc);
    }

    // h_{t-1} @ H^T : A (split) from LDS, B_hi persistent, B_lo streamed from LDS
#pragma unroll
    for (int c = 0; c < KCH; ++c) {
      const int base = lo16 * HSTR + c * 32;
      V16 ah, al, bl;
      ah.h[0] = *(const v8bf*)&hhi[base + ka];
      ah.h[1] = *(const v8bf*)&hhi[base + 16 + ka];
      al.h[0] = *(const v8bf*)&hlo[base + ka];
      al.h[1] = *(const v8bf*)&hlo[base + 16 + ka];
      bl.h[0] = *(const v8bf*)&Hlo[jrow + c * 32 + ka];
      bl.h[1] = *(const v8bf*)&Hlo[jrow + c * 32 + 16 + ka];
      acc = WMMA(ah.v, Hbh[c], acc);   // Ah*Bh
      acc = WMMA(al.v, Hbh[c], acc);   // Al*Bh
      acc = WMMA(ah.v, bl.v,  acc);    // Ah*Bl
    }

    __syncthreads();                  // WAR: all reads of h done before rewrite

#pragma unroll
    for (int r = 0; r < 8; ++r) nh[r] = TANHF(acc[r] + bj);
#pragma unroll
    for (int r = 0; r < 8; ++r) {
      int m = r + hi * 8;             // C/D layout: VGPR r -> M = r (+8 for hi lanes)
      __bf16 vh_ = (__bf16)nh[r];
      hhi[m * HSTR + j] = vh_;
      hlo[m * HSTR + j] = (__bf16)(nh[r] - (float)vh_);
    }

    asm volatile("s_wait_asynccnt 0x0" ::: "memory");  // next x tile landed
    __syncthreads();                  // RAW: h (and x) visible for next step
  }

  // ---- publish fp32 h_T (peeled out of the loop: nh still live in registers)
#pragma unroll
  for (int r = 0; r < 8; ++r) hf[(r + hi * 8) * FSTR + j] = nh[r];
  __syncthreads();

  // ---- readout: out = h_T @ A^T + c  (tiny: 160 dot-products, fp32 VALU)
  if (tid < 16 * OUT_DIM) {
    const int m = tid / OUT_DIM;
    const int o = tid % OUT_DIM;
    float s = cv[o];
    for (int k = 0; k < HID; ++k) s += hf[m * FSTR + k] * A[o * HID + k];
    out[(b0 + m) * OUT_DIM + o] = s;
  }
}

extern "C" void kernel_launch(void* const* d_in, const int* in_sizes, int n_in,
                              void* d_out, int out_size, void* d_ws, size_t ws_size,
                              hipStream_t stream) {
  const float* x = (const float*)d_in[0];
  const float* H = (const float*)d_in[1];
  const float* U = (const float*)d_in[2];
  const float* A = (const float*)d_in[3];
  const float* b = (const float*)d_in[4];
  const float* c = (const float*)d_in[5];
  float* out = (float*)d_out;
  (void)in_sizes; (void)n_in; (void)out_size; (void)d_ws; (void)ws_size;
  rnn_persist<<<dim3(BATCH / 16), dim3(416), 0, stream>>>(x, H, U, A, b, c, out);
}
